// HgpslPool_20839181320909
// MI455X (gfx1250) — matrix operand order repaired and checked by reading, not compiled, not run on hardware
//
#include <hip/hip_runtime.h>
#include <hip/hip_bf16.h>
#include <stdint.h>

#define DVAL   128
#define SLOPE  0.2f
#define LAMBV  1.0f
#define NSORT  8192
#define ESLOTS 64
#define NCHUNK 8
#define ONEONE 0x3c003c00u   // packed f16 {1.0, 1.0}

typedef __attribute__((ext_vector_type(16))) _Float16 v16h;
typedef __attribute__((ext_vector_type(8)))  float    v8f;

union U16H { v16h h; unsigned u[8]; };

__device__ __forceinline__ float wredsum(float v){
#pragma unroll
  for (int o=16;o>0;o>>=1) v += __shfl_xor(v,o,32);
  return v;
}
__device__ __forceinline__ float wredmax(float v){
#pragma unroll
  for (int o=16;o>0;o>>=1) v = fmaxf(v,__shfl_xor(v,o,32));
  return v;
}
__device__ __forceinline__ float lrelu(float x){ return x>0.f? x : SLOPE*x; }

__global__ void k_zero(float* p, long n){
  long i = (long)blockIdx.x*blockDim.x + threadIdx.x;
  long stride = (long)gridDim.x*blockDim.x;
  for(; i<n; i+=stride) p[i]=0.f;
}

// logmap0: xtan = arctanh(||x||)/||x|| * x   (norm clipped to [1e-15, 1-1e-5])
__global__ void k_logmap(const float* __restrict__ x, float* __restrict__ xtan){
  int i=blockIdx.x, d=threadIdx.x;
  float v = x[(long)i*DVAL+d];
  float ss = wredsum(v*v);
  __shared__ float sred[4]; __shared__ float rbc;
  int wid=threadIdx.x>>5, lane=threadIdx.x&31;
  if(!lane) sred[wid]=ss;
  __syncthreads();
  if(threadIdx.x==0){
    float n2=sred[0]+sred[1]+sred[2]+sred[3];
    float nr=sqrtf(n2);
    nr=fminf(fmaxf(nr,1e-15f),1.0f-1e-5f);
    rbc=0.5f*logf((1.f+nr)/(1.f-nr))/nr;   // arctanh(nr)/nr
  }
  __syncthreads();
  xtan[(long)i*DVAL+d]=rbc*v;
}

__global__ void k_deg(const int* __restrict__ row, const float* __restrict__ ea,
                      float* deg, int E){
  int e=blockIdx.x*blockDim.x+threadIdx.x;
  if(e<E) atomicAdd(&deg[row[e]], ea[e]);
}

// acc[col] += -dinv[row]*ea*dinv[col] * xtan[row]
__global__ void k_scatter(const int* __restrict__ row, const int* __restrict__ col,
                          const float* __restrict__ ea, const float* __restrict__ deg,
                          const float* __restrict__ xtan, float* acc){
  int e=blockIdx.x, d=threadIdx.x;
  int r=row[e], c=col[e];
  float dr=deg[r], dc=deg[c];
  float coef = -(dr>0.f? rsqrtf(dr):0.f) * ea[e] * (dc>0.f? rsqrtf(dc):0.f);
  atomicAdd(&acc[(long)c*DVAL+d], coef*xtan[(long)r*DVAL+d]);
}

__global__ void k_score(const float* __restrict__ xtan, const float* __restrict__ acc,
                        float* score){
  int i=blockIdx.x, d=threadIdx.x;
  float v = fabsf(xtan[(long)i*DVAL+d] + acc[(long)i*DVAL+d]);
  float ss = wredsum(v);
  __shared__ float sred[4];
  int wid=threadIdx.x>>5, lane=threadIdx.x&31;
  if(!lane) sred[wid]=ss;
  __syncthreads();
  if(threadIdx.x==0) score[i]=sred[0]+sred[1]+sred[2]+sred[3];
}

// full bitonic sort of (score, idx) in 64KB LDS; descending, ties -> lower index first
__global__ __launch_bounds__(1024) void k_sort(const float* __restrict__ score,
                                               int N, int K, int* perm, int* mapping){
  __shared__ float skey[NSORT];
  __shared__ int   sidx[NSORT];
  for(int t=threadIdx.x;t<NSORT;t+=blockDim.x){
    if(t<N){skey[t]=score[t]; sidx[t]=t;}
    else   {skey[t]=-1e30f;   sidx[t]=0x7fffffff;}
  }
  __syncthreads();
  for(unsigned k=2;k<=NSORT;k<<=1){
    for(unsigned j=k>>1;j>0;j>>=1){
      for(unsigned t=threadIdx.x;t<NSORT;t+=blockDim.x){
        unsigned p=t^j;
        if(p>t){
          float ka=skey[t], kb=skey[p];
          int ia=sidx[t], ib=sidx[p];
          bool aFirst = (ka>kb)||(ka==kb && ia<ib);   // a precedes b in desc order
          bool descBlock = ((t&k)==0u);
          bool doSwap = descBlock ? !aFirst : aFirst;
          if(doSwap){ skey[t]=kb; skey[p]=ka; sidx[t]=ib; sidx[p]=ia; }
        }
      }
      __syncthreads();
    }
  }
  for(int t=threadIdx.x;t<N;t+=blockDim.x){
    int node=sidx[t];
    if(t<K) perm[t]=node;
    mapping[node] = (t<K)? t : -1;
  }
}

// gather x_sel -> d_out; s=x_sel.att_src, t=x_sel.att_dst; packed Dekker f16 splits
__global__ void k_gather(const float* __restrict__ x, const float* __restrict__ att,
                         const int* __restrict__ perm, float* xsel,
                         float* s, float* t, unsigned* spk, unsigned* tpk){
  int k=blockIdx.x, d=threadIdx.x;
  int p=perm[k];
  float v=x[(long)p*DVAL+d];
  xsel[(long)k*DVAL+d]=v;
  float ps=wredsum(v*att[d]);
  float pt=wredsum(v*att[DVAL+d]);
  __shared__ float as_[4], at_[4];
  int wid=threadIdx.x>>5, lane=threadIdx.x&31;
  if(!lane){as_[wid]=ps; at_[wid]=pt;}
  __syncthreads();
  if(threadIdx.x==0){
    float S=as_[0]+as_[1]+as_[2]+as_[3];
    float T=at_[0]+at_[1]+at_[2]+at_[3];
    s[k]=S; t[k]=T;
    _Float16 h=(_Float16)S; _Float16 l=(_Float16)(S-(float)h);
    spk[k] = (unsigned)__builtin_bit_cast(unsigned short, h)
           | ((unsigned)__builtin_bit_cast(unsigned short, l)<<16);
    _Float16 g=(_Float16)T; _Float16 gl=(_Float16)(T-(float)g);
    tpk[k] = (unsigned)__builtin_bit_cast(unsigned short, g)
           | ((unsigned)__builtin_bit_cast(unsigned short, gl)<<16);
  }
}

__global__ void k_bucket(const int* __restrict__ row, const int* __restrict__ col,
                         const float* __restrict__ ea, const int* __restrict__ mapping,
                         int* rowcnt, int* rowcols, float* rowvals, int E){
  int e=blockIdx.x*blockDim.x+threadIdx.x;
  if(e>=E) return;
  int r2=mapping[row[e]], c2=mapping[col[e]];
  if(r2>=0 && c2>=0){
    int pos=atomicAdd(&rowcnt[r2],1);
    if(pos<ESLOTS){ rowcols[(long)r2*ESLOTS+pos]=c2; rowvals[(long)r2*ESLOTS+pos]=LAMBV*ea[e]; }
  }
}

// per-row softmax stats with sparse edge corrections
__global__ __launch_bounds__(256) void k_rowstats(const float* __restrict__ s,
                           const float* __restrict__ t, const int* __restrict__ rowcnt,
                           const int* __restrict__ rowcols, const float* __restrict__ rowvals,
                           int K, float* mrow, float* invrow){
  int i=blockIdx.x;
  __shared__ int   ecol[ESLOTS];
  __shared__ float evalv[ESLOTS];
  __shared__ float sred[8];
  __shared__ float bc;
  int cnt=rowcnt[i]; if(cnt>ESLOTS)cnt=ESLOTS;
  for(int e=threadIdx.x;e<cnt;e+=blockDim.x){
    ecol[e]=rowcols[(long)i*ESLOTS+e]; evalv[e]=rowvals[(long)i*ESLOTS+e];
  }
  __syncthreads();
  float si=s[i];
  float lm=-1e30f;
  for(int j=threadIdx.x;j<K;j+=blockDim.x) lm=fmaxf(lm, lrelu(si+t[j]));
  lm=wredmax(lm);
  int wid=threadIdx.x>>5, lane=threadIdx.x&31;
  if(!lane) sred[wid]=lm;
  __syncthreads();
  if(threadIdx.x==0){
    float m=sred[0];
    for(int w=1;w<8;w++) m=fmaxf(m,sred[w]);
    for(int e=0;e<cnt;e++) m=fmaxf(m, lrelu(si+t[ecol[e]])+evalv[e]);
    bc=m;
  }
  __syncthreads();
  float m=bc;
  float lz=0.f;
  for(int j=threadIdx.x;j<K;j+=blockDim.x) lz+=__expf(lrelu(si+t[j])-m);
  lz=wredsum(lz);
  __syncthreads();
  if(!lane) sred[wid]=lz;
  __syncthreads();
  if(threadIdx.x==0){
    float z=0.f;
    for(int w=0;w<8;w++) z+=sred[w];
    for(int e=0;e<cnt;e++){
      float fc=lrelu(si+t[ecol[e]]);
      z+=__expf(fc+evalv[e]-m)-__expf(fc-m);
    }
    mrow[i]=m; invrow[i]=1.f/z;
  }
}

// WMMA tile kernel: 16x16 tile of (s_i + t_j) via rank-2 outer-sum, one write of adj.
// A[m][0..3]={s_hi,s_lo,1,1}; B[0..3][n]={1,1,t_hi,t_lo}; f32 accumulate -> ~f32 exact.
// Operands built branch-free in ALL lanes: hi-half lane writes land in K-slots whose
// counterpart operand entries are zero, so they cancel in the product.
__global__ __launch_bounds__(256) void k_tiles(const unsigned* __restrict__ spk,
                        const unsigned* __restrict__ tpk,
                        const float* __restrict__ mrow, const float* __restrict__ invrow,
                        int K, int numTiles, int tilesPerChunk, float* __restrict__ adj){
  int br=blockIdx.x;
  int wv=threadIdx.x>>5, lane=threadIdx.x&31;
  bool lo = lane<16;
  int n = lane&15;

  U16H A;
#pragma unroll
  for(int q=0;q<8;q++) A.u[q]=0u;
  A.u[0]=lo? spk[min(br*16+n, K-1)] : 0u;   // {s_hi, s_lo} packed
  A.u[1]=lo? ONEONE : 0u;                   // {1, 1}

  // C/D layout: lane<16 -> N=lane, VGPR v -> M=v ; lane>=16 -> N=lane-16, M=v+8
  float mr[8], ir[8];
#pragma unroll
  for(int v=0;v<8;v++){
    int m = lo? v : v+8;
    int gi=min(br*16+m, K-1);
    mr[v]=mrow[gi]; ir[v]=invrow[gi];
  }
  int rowFull = ((br+1)*16 <= K) ? 1 : 0;
  int fullColTiles = K>>4;
  int nwv = blockDim.x>>5;
  int bcBeg = blockIdx.y*tilesPerChunk;
  int bcEnd = min(numTiles, bcBeg+tilesPerChunk);
  size_t rowOff0 = (size_t)(br*16 + (lo?0:8))*(size_t)K;  // row of c[0] for this lane
  for(int bc=bcBeg+wv; bc<bcEnd; bc+=nwv){
    U16H B;
#pragma unroll
    for(int q=0;q<8;q++) B.u[q]=0u;
    B.u[0]=ONEONE;                       // {1, 1}
    B.u[1]=tpk[min(bc*16+n, K-1)];       // {t_hi, t_lo} packed
    v8f c={};
    c=__builtin_amdgcn_wmma_f32_16x16x32_f16(false,A.h,false,B.h,(short)0,c,false,false);
    int gcol = bc*16+n;
    int fast = __builtin_amdgcn_readfirstlane(rowFull & (int)(bc < fullColTiles));
    if(fast){                              // scalar-uniform fast path
      float* pbase = adj + rowOff0 + gcol;
#pragma unroll
      for(int v=0;v<8;v++){
        float f=lrelu(c[v]);
        pbase[(size_t)v*K]=__expf(f-mr[v])*ir[v];
      }
    } else {                               // edge tiles only
      if(gcol<K){
#pragma unroll
        for(int v=0;v<8;v++){
          int m = lo? v : v+8;
          int grow=br*16+m;
          if(grow<K){
            float f=lrelu(c[v]);
            adj[(size_t)grow*K+gcol]=__expf(f-mr[v])*ir[v];
          }
        }
      }
    }
  }
}

// overwrite the <=E edge positions with the edge-corrected softmax value
__global__ void k_fixup(const int* __restrict__ row, const int* __restrict__ col,
                        const float* __restrict__ ea, const int* __restrict__ mapping,
                        const float* __restrict__ s, const float* __restrict__ t,
                        const float* __restrict__ mrow, const float* __restrict__ invrow,
                        int K, int E, float* adj){
  int e=blockIdx.x*blockDim.x+threadIdx.x;
  if(e>=E) return;
  int r2=mapping[row[e]], c2=mapping[col[e]];
  if(r2<0||c2<0) return;
  float f=lrelu(s[r2]+t[c2])+LAMBV*ea[e];
  adj[(size_t)r2*K+c2]=__expf(f-mrow[r2])*invrow[r2];
}

extern "C" void kernel_launch(void* const* d_in, const int* in_sizes, int n_in,
                              void* d_out, int out_size, void* d_ws, size_t ws_size,
                              hipStream_t stream){
  (void)n_in; (void)out_size; (void)ws_size;
  const float* x   = (const float*)d_in[0];
  const int*   ei  = (const int*)d_in[1];
  const float* ea  = (const float*)d_in[2];
  const float* att = (const float*)d_in[3];
  int N = in_sizes[0]/DVAL;
  int E = in_sizes[2];
  const int* row = ei;
  const int* col = ei + E;
  int K = (4*N+4)/5;                 // ceil(0.8*N)
  int numTiles=(K+15)/16;
  int tilesPerChunk=(numTiles+NCHUNK-1)/NCHUNK;

  char* p=(char*)d_ws;
  auto carve=[&](size_t bytes)->char*{
    char* r=(char*)(((uintptr_t)p+255)&~(uintptr_t)255);
    p=r+bytes; return r;
  };
  float*    xtan   =(float*)   carve((size_t)N*DVAL*4);
  float*    acc    =(float*)   carve((size_t)N*DVAL*4);
  float*    deg    =(float*)   carve((size_t)N*4);
  float*    score  =(float*)   carve((size_t)N*4);
  int*      perm   =(int*)     carve((size_t)K*4);
  int*      mapping=(int*)     carve((size_t)N*4);
  float*    sArr   =(float*)   carve((size_t)K*4);
  float*    tArr   =(float*)   carve((size_t)K*4);
  unsigned* spk    =(unsigned*)carve((size_t)K*4);
  unsigned* tpk    =(unsigned*)carve((size_t)K*4);
  float*    mrow   =(float*)   carve((size_t)K*4);
  float*    invrow =(float*)   carve((size_t)K*4);
  int*      rowcnt =(int*)     carve((size_t)K*4);
  int*      rowcols=(int*)     carve((size_t)K*ESLOTS*4);
  float*    rowvals=(float*)   carve((size_t)K*ESLOTS*4);

  float* xsel=(float*)d_out;
  float* adj =(float*)d_out + (size_t)K*DVAL;

  k_zero<<<512,256,0,stream>>>(acc,(long)N*DVAL);
  k_zero<<<64,256,0,stream>>>(deg,N);
  k_zero<<<64,256,0,stream>>>((float*)rowcnt,K);
  k_logmap<<<N,DVAL,0,stream>>>(x,xtan);
  k_deg<<<(E+255)/256,256,0,stream>>>(row,ea,deg,E);
  k_scatter<<<E,DVAL,0,stream>>>(row,col,ea,deg,xtan,acc);
  k_score<<<N,DVAL,0,stream>>>(xtan,acc,score);
  k_sort<<<1,1024,0,stream>>>(score,N,K,perm,mapping);
  k_gather<<<K,DVAL,0,stream>>>(x,att,perm,xsel,sArr,tArr,spk,tpk);
  k_bucket<<<(E+255)/256,256,0,stream>>>(row,col,ea,mapping,rowcnt,rowcols,rowvals,E);
  k_rowstats<<<K,256,0,stream>>>(sArr,tArr,rowcnt,rowcols,rowvals,K,mrow,invrow);
  dim3 tg(numTiles, NCHUNK);
  k_tiles<<<tg,256,0,stream>>>(spk,tpk,mrow,invrow,K,numTiles,tilesPerChunk,adj);
  k_fixup<<<(E+255)/256,256,0,stream>>>(row,col,ea,mapping,sArr,tArr,mrow,invrow,K,E,adj);
}
